// TriangleAttention_26783416058538
// MI455X (gfx1250) — matrix operand order repaired and checked by reading, not compiled, not run on hardware
//
#include <hip/hip_runtime.h>
#include <hip/hip_bf16.h>

// ---------------------------------------------------------------------------
// Triangle attention (outgoing) for gfx1250 / MI455X.
// All matmuls via v_wmma_f32_16x16x32_f16 (wave32 WMMA).
// Pipeline: weight-swizzle -> layernorm -> QKV/bias proj -> attention -> out.
// ---------------------------------------------------------------------------

typedef _Float16 v16h __attribute__((ext_vector_type(16)));
typedef _Float16 v8h  __attribute__((ext_vector_type(8)));
typedef _Float16 v4h  __attribute__((ext_vector_type(4)));
typedef __fp16   pk2  __attribute__((ext_vector_type(2)));   // cvt_pkrtz result
typedef float    v8f  __attribute__((ext_vector_type(8)));

#define B_   2
#define L_   256
#define C_   128
#define H_   8
#define DH_  16
#define RROWS (B_ * L_ * L_)   // 131072 rows of width 128

union V16 {
  v16h v;
  v8h  h2[2];
  _Float16 e[16];
};

__device__ __forceinline__ v8f wmma16(v16h a, v16h b, v8f c) {
  // D = A(16x32 f16) x B(32x16 f16) + C(16x16 f32)
  return __builtin_amdgcn_wmma_f32_16x16x32_f16(false, a, false, b,
                                                (short)0, c, false, false);
}

// ---------------------------------------------------------------------------
// Kernel 0: convert f32 weights into f16 WMMA B-fragments.
// B-layout (ISA 7.12.2, 16-bit B 32x16): slot j -> K = (j%8) + 8*(lane/16)
//                                        + 16*(j/8), N = lane%16.
// Fragment storage: frag[(kt*8+nt)*32*16 + lane*16 + j]  (32B/lane contiguous)
// ---------------------------------------------------------------------------
__global__ __launch_bounds__(32) void swz_kernel(
    const float* __restrict__ Wq, const float* __restrict__ Wk,
    const float* __restrict__ Wv, const float* __restrict__ Wo,
    const float* __restrict__ Wg, const float* __restrict__ Wb,
    _Float16* __restrict__ oQ, _Float16* __restrict__ oK,
    _Float16* __restrict__ oV, _Float16* __restrict__ oO,
    _Float16* __restrict__ oG, _Float16* __restrict__ oB) {
  const int lane = threadIdx.x;
  const int t    = blockIdx.x;          // tile = kt*8 + nt
  const int wsel = blockIdx.y;          // which weight
  const int kt = t >> 3, nt = t & 7;
  if (wsel < 5) {
    const float* Ws[5] = {Wq, Wk, Wv, Wo, Wg};
    _Float16*    Ds[5] = {oQ, oK, oV, oO, oG};
    const float* W = Ws[wsel];
    _Float16*    D = Ds[wsel];
#pragma unroll
    for (int j = 0; j < 16; ++j) {
      const int krow = kt * 32 + (j & 7) + ((lane >> 4) << 3) + ((j >> 3) << 4);
      D[(t * 32 + lane) * 16 + j] =
          (_Float16)W[krow * C_ + nt * 16 + (lane & 15)];
    }
  } else {                               // Wb: 128x8, pad N to 16 with zeros
    if (nt != 0) return;
    const int col = lane & 15;
#pragma unroll
    for (int j = 0; j < 16; ++j) {
      const int krow = kt * 32 + (j & 7) + ((lane >> 4) << 3) + ((j >> 3) << 4);
      const float val = (col < H_) ? Wb[krow * H_ + col] : 0.0f;
      oB[(kt * 32 + lane) * 16 + j] = (_Float16)val;
    }
  }
}

// ---------------------------------------------------------------------------
// Kernel 1: LayerNorm over C=128, one wave32 per row, f16 output.
// ---------------------------------------------------------------------------
__global__ __launch_bounds__(256) void ln_kernel(
    const float* __restrict__ z, const float* __restrict__ gamma,
    const float* __restrict__ beta, _Float16* __restrict__ zn) {
  const int lane = threadIdx.x & 31;
  const int wave = threadIdx.x >> 5;
  const long row = (long)blockIdx.x * 8 + wave;
  const float4 x = *(const float4*)(z + row * C_ + lane * 4);
  float s1 = x.x + x.y + x.z + x.w;
  float s2 = x.x * x.x + x.y * x.y + x.z * x.z + x.w * x.w;
#pragma unroll
  for (int o = 16; o > 0; o >>= 1) {
    s1 += __shfl_xor(s1, o);
    s2 += __shfl_xor(s2, o);
  }
  const float mu  = s1 * (1.0f / 128.0f);
  const float var = s2 * (1.0f / 128.0f) - mu * mu;
  const float rs  = rsqrtf(var + 1e-5f);
  const int c = lane * 4;
  v4h y;
  y[0] = (_Float16)((x.x - mu) * rs * gamma[c + 0] + beta[c + 0]);
  y[1] = (_Float16)((x.y - mu) * rs * gamma[c + 1] + beta[c + 1]);
  y[2] = (_Float16)((x.z - mu) * rs * gamma[c + 2] + beta[c + 2]);
  y[3] = (_Float16)((x.w - mu) * rs * gamma[c + 3] + beta[c + 3]);
  *(v4h*)&zn[row * C_ + c] = y;
}

// ---------------------------------------------------------------------------
// 16x128 GEMM strip with double-buffered B-fragments and LDS-transposed
// b128 epilogue stores. stg = per-wave 256-half staging tile (row-major 16x16).
// C/D layout: reg r -> M = r + 8*(lane/16), N = lane%16.
// ---------------------------------------------------------------------------
__device__ __forceinline__ void gemm16x128(
    const V16 a[4], const _Float16* __restrict__ w,
    const float* __restrict__ bias, _Float16* __restrict__ dst,
    _Float16* stg, long m0, int lane) {
  const int hl = lane >> 4;
  v16h bf[4];
#pragma unroll
  for (int kt = 0; kt < 4; ++kt)
    bf[kt] = *(const v16h*)&w[((kt * 8 + 0) * 32 + lane) * 16];
#pragma unroll
  for (int nt = 0; nt < 8; ++nt) {
    v16h cur[4];
#pragma unroll
    for (int kt = 0; kt < 4; ++kt) cur[kt] = bf[kt];
    if (nt < 7) {  // prefetch next N-tile's B-frags before the WMMA chain
#pragma unroll
      for (int kt = 0; kt < 4; ++kt)
        bf[kt] = *(const v16h*)&w[((kt * 8 + nt + 1) * 32 + lane) * 16];
    }
    v8f acc = {};
#pragma unroll
    for (int kt = 0; kt < 4; ++kt) acc = wmma16(a[kt].v, cur[kt], acc);
    const float bv = bias[nt * 16 + (lane & 15)];
#pragma unroll
    for (int r = 0; r < 8; ++r)
      stg[(r + 8 * hl) * 16 + (lane & 15)] = (_Float16)(acc[r] + bv);
    // same-wave LDS transpose (in-order per wave) -> one b128 store per lane
    const v8h row8 = *(const v8h*)&stg[lane * 8];
    *(v8h*)&dst[(m0 + (lane >> 1)) * C_ + nt * 16 + (lane & 1) * 8] = row8;
  }
}

// ---------------------------------------------------------------------------
// Kernel 2: Q/K/V projections + pair-bias head (zn @ Wb -> (B,H,L,L)).
// A-layout (ISA 7.12.2, 16-bit A 16x32): M = lane%16,
//   slot j -> K = (j%8) + 8*(lane/16) + 16*(j/8)  => two 16B loads per frag.
// ---------------------------------------------------------------------------
__global__ __launch_bounds__(256) void proj_kernel(
    const _Float16* __restrict__ zn,
    const _Float16* __restrict__ wQ, const _Float16* __restrict__ wK,
    const _Float16* __restrict__ wV, const _Float16* __restrict__ wB,
    const float* __restrict__ bq, const float* __restrict__ bk,
    const float* __restrict__ bv,
    _Float16* __restrict__ Qf, _Float16* __restrict__ Kf,
    _Float16* __restrict__ Vf, float* __restrict__ biasF) {
  __shared__ __align__(16) _Float16 stg_all[8][256];   // 4 KB staging
  const int lane = threadIdx.x & 31;
  const int wave = threadIdx.x >> 5;
  _Float16* stg = &stg_all[wave][0];
  const long m0   = ((long)blockIdx.x * 8 + wave) * 16;
  const long mrow = m0 + (lane & 15);
  V16 a[4];
#pragma unroll
  for (int kt = 0; kt < 4; ++kt) {
    const _Float16* pz = zn + mrow * C_ + kt * 32 + ((lane >> 4) << 3);
    a[kt].h2[0] = *(const v8h*)pz;          // K = kb + 0..7  (per-lane half)
    a[kt].h2[1] = *(const v8h*)(pz + 16);   // K = kb + 16..23
  }
  gemm16x128(a, wQ, bq, Qf, stg, m0, lane);
  gemm16x128(a, wK, bk, Kf, stg, m0, lane);
  gemm16x128(a, wV, bv, Vf, stg, m0, lane);

  // pair bias: zn @ Wb (no bias vector), scatter to (B,H,Lj,Lk)
  v16h bf[4];
#pragma unroll
  for (int kt = 0; kt < 4; ++kt)
    bf[kt] = *(const v16h*)&wB[(kt * 32 + lane) * 16];
  v8f acc = {};
#pragma unroll
  for (int kt = 0; kt < 4; ++kt) acc = wmma16(a[kt].v, bf[kt], acc);
  const int hcol = lane & 15;
  if (hcol < H_) {
#pragma unroll
    for (int r = 0; r < 8; ++r) {
      const long row = m0 + r + ((lane >> 4) << 3);   // (b, jj, kk) flat
      const int b  = (int)(row >> 16);
      const int jj = (int)((row >> 8) & 255);
      const int kk = (int)(row & 255);
      biasF[(((long)b * H_ + hcol) * L_ + jj) * L_ + kk] = acc[r];
    }
  }
}

// ---------------------------------------------------------------------------
// Kernel 3: attention per (b,h,i). 64 threads = 2 waves; each wave owns one
// 16-row j-tile per iteration (8 iterations).
//  - K B-frags read straight from global (16B/lane, L0-resident across iters)
//  - row max tracked in registers during the score phase (shfl-tree reduce)
//  - single exp pass packs f16 pairs (v_cvt_pk_rtz) directly into A-fragment
//    layout in LDS; 1/sum applied to the f32 output accumulators instead
// ---------------------------------------------------------------------------
__global__ __launch_bounds__(64) void attn_kernel(
    const _Float16* __restrict__ Qf, const _Float16* __restrict__ Kf,
    const _Float16* __restrict__ Vf, const float* __restrict__ biasF,
    _Float16* __restrict__ Of) {
  __shared__ __align__(32) _Float16 vfrag[8 * 512];       //  8 KB: V B-frags
  __shared__ float probs[2][16 * 256];                    // 32 KB: raw scores
  __shared__ __align__(32) _Float16 pfrag[2][8 * 512];    // 16 KB: exp A-frags
  __shared__ __align__(32) float rowmax[2][16];
  __shared__ __align__(32) float rowinv[2][16];
  __shared__ __align__(16) _Float16 ostg[2][256];         //  1 KB: O staging

  const int x = blockIdx.x;
  const int i = x & 255;
  const int h = (x >> 8) & 7;
  const int b = x >> 11;
  const long rowBase = ((long)(b * L_ + i)) * L_;
  const int tid = threadIdx.x, lane = tid & 31, w = tid >> 5;
  const int hl = lane >> 4;

  // stage V as B-fragments: slot j -> K(row)=ks*32+(j%8)+8*(lane/16)+16*(j/8), N=d=lane%16
  for (int idx = tid; idx < 4096; idx += 64) {
    const int ks = idx >> 9, l = (idx >> 4) & 31, j = idx & 15;
    const int kk = ks * 32 + (j & 7) + ((l >> 4) << 3) + ((j >> 3) << 4);
    vfrag[idx] = Vf[(rowBase + kk) * C_ + h * DH_ + (l & 15)];
  }
  __syncthreads();

  const float scale = 0.25f;  // 1/sqrt(16)
  for (int it = 0; it < 8; ++it) {
    const int jt = it * 2 + w;
    // Q A-fragment (d padded 16->32: upper slots zero)
    V16 qa;
    const int jrow = jt * 16 + (lane & 15);
    qa.h2[0] = *(const v8h*)&Qf[(rowBase + jrow) * C_ + h * DH_ + (hl << 3)];
#pragma unroll
    for (int j = 8; j < 16; ++j) qa.e[j] = (_Float16)0.0f;

    // ---- scores strip 16 x 256: one WMMA per k-tile; track row max in regs
    float m[8];
#pragma unroll
    for (int r = 0; r < 8; ++r) m[r] = -3.0e38f;
    const int jl = hl << 3;
    for (int kt = 0; kt < 16; ++kt) {
      V16 kb;
      {  // B-frag: slots 0..7 -> d = 8*(lane/16) + 0..7, slots 8..15 -> d>=16 pad
        kb.h2[0] = *(const v8h*)&Kf[(rowBase + kt * 16 + (lane & 15)) * C_ + h * DH_ + (hl << 3)];
#pragma unroll
        for (int j = 8; j < 16; ++j) kb.e[j] = (_Float16)0.0f;  // d>=16 pad
      }
      v8f c = {};
      c = wmma16(qa.v, kb.v, c);
      const int col = kt * 16 + (lane & 15);
      const float* bp =
          &biasF[(((long)(b * H_ + h)) * L_ + jt * 16) * L_ + col];
#pragma unroll
      for (int r = 0; r < 8; ++r) {
        const float sv = c[r] * scale + bp[(r + jl) * 256];
        probs[w][(r + jl) * 256 + col] = sv;
        m[r] = fmaxf(m[r], sv);
      }
    }
    // cross-lane row max within each 16-lane half
#pragma unroll
    for (int off = 1; off < 16; off <<= 1)
#pragma unroll
      for (int r = 0; r < 8; ++r) m[r] = fmaxf(m[r], __shfl_xor(m[r], off));
    if ((lane & 15) == 0) {
#pragma unroll
      for (int r = 0; r < 8; ++r) rowmax[w][jl + r] = m[r];
    }
    __syncthreads();

    // ---- single exp pass: pack f16 pairs straight into A-fragment layout
    {
      const int row = lane & 15, ch = hl;
      const float mx = rowmax[w][row];
      const float* pr = &probs[w][row * 256 + (ch << 7)];
      float s = 0.0f;
#pragma unroll 4
      for (int c0 = 0; c0 < 128; c0 += 2) {
        const float2 pv = *(const float2*)&pr[c0];
        const float e0 = __expf(pv.x - mx);
        const float e1 = __expf(pv.y - mx);
        s += e0 + e1;
        const pk2 pk = __builtin_amdgcn_cvt_pkrtz(e0, e1);
        const int k = (ch << 7) + c0;       // even => pair contiguous in frag
        const int ks = k >> 5;
        const int lanep = row + (((k >> 3) & 1) << 4);
        const int j = (k & 7) + (((k >> 4) & 1) << 3);
        *(pk2*)&pfrag[w][ks * 512 + lanep * 16 + j] = pk;
      }
      s += __shfl_xor(s, 16);
      if (ch == 0) rowinv[w][row] = 1.0f / s;
    }
    __syncthreads();

    // ---- out = exp(scores) @ V  (8 full-K WMMA steps), then scale by 1/sum
    v8f acc = {};
#pragma unroll
    for (int ks = 0; ks < 8; ++ks) {
      const v16h pa = *(const v16h*)&pfrag[w][ks * 512 + lane * 16];
      const v16h vb = *(const v16h*)&vfrag[ks * 512 + lane * 16];
      acc = wmma16(pa, vb, acc);
    }
    const float4 i0 = *(const float4*)&rowinv[w][hl * 8];
    const float4 i1 = *(const float4*)&rowinv[w][hl * 8 + 4];
    const float invv[8] = {i0.x, i0.y, i0.z, i0.w, i1.x, i1.y, i1.z, i1.w};
#pragma unroll
    for (int r = 0; r < 8; ++r)
      ostg[w][(r + 8 * hl) * 16 + (lane & 15)] = (_Float16)(acc[r] * invv[r]);
    // same-wave LDS transpose -> one b128 store per lane
    const v8h orow = *(const v8h*)&ostg[w][lane * 8];
    *(v8h*)&Of[(rowBase + jt * 16 + (lane >> 1)) * C_ + h * DH_ +
               (lane & 1) * 8] = orow;
  }
}

// ---------------------------------------------------------------------------
// Kernel 4: out = z + sigmoid(zn@Wg + bg) * (O@Wo + bo)   (two fused GEMMs)
// ---------------------------------------------------------------------------
__global__ __launch_bounds__(256) void final_kernel(
    const _Float16* __restrict__ Of, const _Float16* __restrict__ zn,
    const _Float16* __restrict__ wO, const _Float16* __restrict__ wG,
    const float* __restrict__ bo, const float* __restrict__ bg,
    const float* __restrict__ z, float* __restrict__ out) {
  const int lane = threadIdx.x & 31;
  const int wave = threadIdx.x >> 5;
  const long m0   = ((long)blockIdx.x * 8 + wave) * 16;
  const long mrow = m0 + (lane & 15);
  V16 aO[4], aZ[4];
#pragma unroll
  for (int kt = 0; kt < 4; ++kt) {
    const _Float16* po = Of + mrow * C_ + kt * 32 + ((lane >> 4) << 3);
    const _Float16* pz = zn + mrow * C_ + kt * 32 + ((lane >> 4) << 3);
    aO[kt].h2[0] = *(const v8h*)po;
    aO[kt].h2[1] = *(const v8h*)(po + 16);
    aZ[kt].h2[0] = *(const v8h*)pz;
    aZ[kt].h2[1] = *(const v8h*)(pz + 16);
  }
#pragma unroll
  for (int nt = 0; nt < 8; ++nt) {
    v16h bo_f[4], bg_f[4];   // 16-load clause ahead of the 8-WMMA chain
#pragma unroll
    for (int kt = 0; kt < 4; ++kt) {
      bo_f[kt] = *(const v16h*)&wO[((kt * 8 + nt) * 32 + lane) * 16];
      bg_f[kt] = *(const v16h*)&wG[((kt * 8 + nt) * 32 + lane) * 16];
    }
    v8f co = {}, cg = {};
#pragma unroll
    for (int kt = 0; kt < 4; ++kt) {
      co = wmma16(aO[kt].v, bo_f[kt], co);
      cg = wmma16(aZ[kt].v, bg_f[kt], cg);
    }
    const int col = nt * 16 + (lane & 15);
    const float bov = bo[col], bgv = bg[col];
#pragma unroll
    for (int r = 0; r < 8; ++r) {
      const long row = m0 + r + ((lane >> 4) << 3);
      const float g = 1.0f / (1.0f + __expf(-(cg[r] + bgv)));
      out[row * C_ + col] = z[row * C_ + col] + g * (co[r] + bov);
    }
  }
}

// ---------------------------------------------------------------------------
extern "C" void kernel_launch(void* const* d_in, const int* in_sizes, int n_in,
                              void* d_out, int out_size, void* d_ws,
                              size_t ws_size, hipStream_t stream) {
  const float* z     = (const float*)d_in[0];
  const float* gamma = (const float*)d_in[1];
  const float* beta  = (const float*)d_in[2];
  const float* Wb    = (const float*)d_in[3];
  const float* Wq    = (const float*)d_in[4];
  const float* bq    = (const float*)d_in[5];
  const float* Wk    = (const float*)d_in[6];
  const float* bk    = (const float*)d_in[7];
  const float* Wv    = (const float*)d_in[8];
  const float* bv    = (const float*)d_in[9];
  const float* Wo    = (const float*)d_in[10];
  const float* bo    = (const float*)d_in[11];
  const float* Wg    = (const float*)d_in[12];
  const float* bg    = (const float*)d_in[13];
  float* out = (float*)d_out;

  // workspace layout (bytes)
  char* ws = (char*)d_ws;
  const size_t BIG = (size_t)RROWS * C_ * sizeof(_Float16);  // 33.5 MB each
  _Float16* zn  = (_Float16*)(ws + 0 * BIG);
  _Float16* Qf  = (_Float16*)(ws + 1 * BIG);
  _Float16* Kf  = (_Float16*)(ws + 2 * BIG);
  _Float16* Vf  = (_Float16*)(ws + 3 * BIG);
  _Float16* Of  = (_Float16*)(ws + 4 * BIG);
  float*    biasF = (float*)(ws + 5 * BIG);                 // 4.19 MB
  char* wb = ws + 5 * BIG + (size_t)B_ * H_ * L_ * L_ * sizeof(float);
  const size_t WSZ = 4 * 8 * 32 * 16 * sizeof(_Float16);    // 32 KB per weight
  _Float16* wQw = (_Float16*)(wb + 0 * WSZ);
  _Float16* wKw = (_Float16*)(wb + 1 * WSZ);
  _Float16* wVw = (_Float16*)(wb + 2 * WSZ);
  _Float16* wOw = (_Float16*)(wb + 3 * WSZ);
  _Float16* wGw = (_Float16*)(wb + 4 * WSZ);
  _Float16* wBw = (_Float16*)(wb + 5 * WSZ);                // 4 KB used

  swz_kernel<<<dim3(32, 6), 32, 0, stream>>>(Wq, Wk, Wv, Wo, Wg, Wb,
                                             wQw, wKw, wVw, wOw, wGw, wBw);
  ln_kernel<<<RROWS / 8, 256, 0, stream>>>(z, gamma, beta, zn);
  proj_kernel<<<RROWS / 128, 256, 0, stream>>>(zn, wQw, wKw, wVw, wBw,
                                               bq, bk, bv, Qf, Kf, Vf, biasF);
  attn_kernel<<<B_ * H_ * L_, 64, 0, stream>>>(Qf, Kf, Vf, biasF, Of);
  final_kernel<<<RROWS / 128, 256, 0, stream>>>(Of, zn, wOw, wGw, bo, bg, z,
                                                out);
}
